// BimodalAttentionSet_29858612642056
// MI455X (gfx1250) — matrix operand order repaired and checked by reading, not compiled, not run on hardware
//
#include <hip/hip_runtime.h>
#include <hip/hip_fp16.h>

// BimodalAttentionSet for MI455X (gfx1250, wave32).
// One 256-thread block (8 waves) per sample. Per pair (p,q):
//   - block-uniform nonzero flags -> identity path (just 0.5*x) or softmax path
//   - softmax path: thread t owns row r=t; single exp pass computes
//     Z_r, S_r and stores E^T (f16, max-subtracted so E<=1) into LDS.
//   - row-side output v_r = S_r/Z_r -> acc[q]
//   - column-side output u = E^T * g (g = 0.5*x_p/Z) done with
//     v_wmma_f32_16x16x32_f16: A tiles = E^T slabs from LDS, B = g
//     replicated across N. D is N-uniform => u extracted from lanes 0/16.
// Output [3, B, 256] written exactly once per element (no atomics).

typedef __attribute__((ext_vector_type(16))) _Float16 v16h;
typedef __attribute__((ext_vector_type(8)))  _Float16 v8h;
typedef __attribute__((ext_vector_type(8)))  float    v8f;

#define DDIM 256
#define NTHREADS 256

// LDS layout (dynamic):
//   ET   : 256*256 f16  (E^T: ET[c*256 + r])          131072 B
//   bqs  : 256 f32   (x_q row broadcast)                1024 B
//   red  : 256 f32   (reduction scratch)                1024 B
//   ubuf : 256 f32   (u spill from WMMA D)              1024 B
//   g16  : 256 f16   (0.5 * x_p / Z, 16B aligned)        512 B
#define SMEM_BYTES (131072 + 1024 + 1024 + 1024 + 512)

union V16U { v16h v; v8h h[2]; };

__device__ __forceinline__ float blockMax(float v, float* red, int t) {
    __syncthreads();                 // protect scratch reuse
    red[t] = v;
    __syncthreads();
#pragma unroll
    for (int s = 128; s > 0; s >>= 1) {
        if (t < s) { float o = red[t + s]; if (o > red[t]) red[t] = o; }
        __syncthreads();
    }
    return red[0];
}

__global__ __launch_bounds__(NTHREADS)
void bimodal_attention_set_kernel(const float* __restrict__ x0,
                                  const float* __restrict__ x1,
                                  const float* __restrict__ x2,
                                  float* __restrict__ out, int B) {
    extern __shared__ char smem[];
    _Float16* ET   = (_Float16*)smem;                       // [256*256]
    float*    bqs  = (float*)(smem + 131072);               // [256]
    float*    red  = bqs + 256;                             // [256]
    float*    ubuf = red + 256;                             // [256]
    _Float16* g16  = (_Float16*)(ubuf + 256);               // [256], 16B aligned

    const int b    = blockIdx.x;
    const int t    = threadIdx.x;          // 0..255, also row index r
    const int lane = t & 31;               // wave32
    const int wave = t >> 5;               // 0..7
    const int half = lane >> 4;            // 0: lanes 0-15, 1: lanes 16-31

    const float* xs0 = x0 + (size_t)b * DDIM;
    const float* xs1 = x1 + (size_t)b * DDIM;
    const float* xs2 = x2 + (size_t)b * DDIM;

    float xv[3];
    xv[0] = xs0[t]; xv[1] = xs1[t]; xv[2] = xs2[t];

    // per-modality any-nonzero flags (block-uniform)
    bool nz[3];
#pragma unroll
    for (int i = 0; i < 3; ++i)
        nz[i] = (blockMax(__builtin_fabsf(xv[i]), red, t) != 0.0f);

    float acc[3] = {0.0f, 0.0f, 0.0f};
    const int PP[3] = {0, 0, 1};
    const int QQ[3] = {1, 2, 2};

#pragma unroll
    for (int pr = 0; pr < 3; ++pr) {
        const int p = PP[pr], q = QQ[pr];
        if (nz[p] || nz[q]) {
            // identity attention: each pair contributes 0.5*x_i to result_i
            acc[p] += 0.5f * xv[p];
            acc[q] += 0.5f * xv[q];
        } else {
            // ---- softmax path (block-uniform branch: EXEC all ones) ----
            const float a = xv[p];                 // x_p[b, r=t]
            __syncthreads();                       // ET/bqs safe to reuse
            bqs[t] = xv[q];
            const float qmax =  blockMax( xv[q], red, t);
            const float qmin = -blockMax(-xv[q], red, t);
            // exact row max of a * x_q[c]
            const float mrow = (a >= 0.0f) ? a * qmax : a * qmin;

            // Phase 1: one exp pass. E[r,c] = exp(a*b_c - mrow) in (0,1].
            float Z = 0.0f, S = 0.0f;
#pragma unroll 4
            for (int c = 0; c < DDIM; ++c) {
                const float bc = bqs[c];
                const float e  = __expf(__builtin_fmaf(a, bc, -mrow));
                Z += e;
                S  = __builtin_fmaf(e, bc, S);
                ET[c * DDIM + t] = (_Float16)e;    // E^T[c][r]
            }
            const float invZ = 1.0f / Z;
            acc[q] += 0.5f * S * invZ;             // row-side matvec
            g16[t]  = (_Float16)(0.5f * a * invZ); // weights for column side
            __syncthreads();

            // Phase 2: u = E^T * g via v_wmma_f32_16x16x32_f16.
            // A[m,k] = E[r0+k, c0+m]  (16-bit A layout: lane half selects K
            //  subsets {0..7,16..23} / {8..15,24..31})
            // B[k,n] = g[r0+k] replicated over n (lane half = K {0..7,16..23}/{8..15,24..31})
            const int cme = lane & 15;             // M within tile = column
#pragma unroll
            for (int gidx = 0; gidx < 2; ++gidx) {
                const int c0 = (wave * 2 + gidx) * 16;
                const _Float16* colp = ET + (c0 + cme) * DDIM;
                v8f dacc = {};
                for (int r0 = 0; r0 < DDIM; r0 += 32) {
                    V16U ua, ub;
                    const int o1 = r0 + (half ? 8  : 0);
                    const int o2 = r0 + (half ? 24 : 16);
                    ua.h[0] = *(const v8h*)(colp + o1);
                    ua.h[1] = *(const v8h*)(colp + o2);
                    const _Float16* gp = g16 + r0 + half * 8;
                    ub.h[0] = *(const v8h*)(gp);
                    ub.h[1] = *(const v8h*)(gp + 16);
                    dacc = __builtin_amdgcn_wmma_f32_16x16x32_f16(
                        /*neg_a=*/false, ua.v, /*neg_b=*/false, ub.v,
                        /*c_mod=*/(short)0, dacc,
                        /*reuse_a=*/false, /*reuse_b=*/false);
                }
                // D is N-uniform; lane 0 holds M=0..7, lane 16 holds M=8..15
                if (cme == 0) {
                    const int base = c0 + half * 8;
#pragma unroll
                    for (int v = 0; v < 8; ++v) ubuf[base + v] = dacc[v];
                }
            }
            __syncthreads();
            acc[p] += ubuf[t];                     // 0.5 folded into g
            __syncthreads();
        }
    }

    const size_t bd = (size_t)b * DDIM + t;
    const size_t stride = (size_t)B * DDIM;
    out[0 * stride + bd] = acc[0];
    out[1 * stride + bd] = acc[1];
    out[2 * stride + bd] = acc[2];
}

extern "C" void kernel_launch(void* const* d_in, const int* in_sizes, int n_in,
                              void* d_out, int out_size, void* d_ws, size_t ws_size,
                              hipStream_t stream) {
    (void)n_in; (void)out_size; (void)d_ws; (void)ws_size;
    const float* x0 = (const float*)d_in[0];
    const float* x1 = (const float*)d_in[1];
    const float* x2 = (const float*)d_in[2];
    float* out = (float*)d_out;
    const int B = in_sizes[0] / DDIM;   // 2048

    // Allow large dynamic LDS (131.5 KB; WGP has 320 KB). Deterministic,
    // not a stream op, safe under graph capture.
    hipFuncSetAttribute((const void*)bimodal_attention_set_kernel,
                        hipFuncAttributeMaxDynamicSharedMemorySize, SMEM_BYTES);

    bimodal_attention_set_kernel<<<B, NTHREADS, SMEM_BYTES, stream>>>(
        x0, x1, x2, out, B);
}
